// ImageTransformer2D_3D_Cross_49435073577551
// MI455X (gfx1250) — matrix operand order repaired and checked by reading, not compiled, not run on hardware
//
#include <hip/hip_runtime.h>
#include <hip/hip_bf16.h>
#include <math.h>

typedef __attribute__((ext_vector_type(16))) _Float16 v16h;
typedef __attribute__((ext_vector_type(8)))  float    v8f;

// ---------------- problem constants ----------------
constexpr int B_ = 2, C_ = 128, N_ = 400, V_ = 6, KV_ = 2816;
constexpr int H_ = 8, FFN_ = 256;
constexpr int VK_ = V_ * KV_;      // 16896
constexpr int BN_ = B_ * N_;       // 800
constexpr int QT_ = N_ / 16;       // 25 query tiles
constexpr int SPLIT_ = 8;          // cross-attn split-K factor (1056/8 = 132 tiles/wave)
constexpr float IMG_X_ = 704.0f / 8.0f;   // 88
constexpr float IMG_Y_ = 256.0f / 8.0f;   // 32
constexpr float LOG_EPS_ = -15.942385f;   // log(FLT_EPSILON)
constexpr float NEG_ = -3.0e38f;

__device__ __forceinline__ float sigm(float x) { return 1.0f / (1.0f + __expf(-x)); }

// =====================================================================
// Generic WMMA GEMM: Y[M,Nn] = act(X[M,K] @ W[K,Nn] + bias)
// One wave (32 lanes) per 16x16 output tile; K stepped by 32.
// M % 16 == 0, K % 32 == 0, Nn % 16 == 0 (true for all call sites).
// =====================================================================
__global__ void wmma_gemm(const float* __restrict__ X, const float* __restrict__ W,
                          const float* __restrict__ bias, float* __restrict__ Y,
                          int M, int K, int Nn, int relu) {
    const int lane = threadIdx.x;
    const int m0 = blockIdx.x * 16;
    const int n0 = blockIdx.y * 16;
    __shared__ float sA[16 * 32];
    __shared__ float sB[32 * 16];
    v8f c = {};
    const int mr = lane & 15, hi = lane >> 4;
    const int ar = lane >> 1, acb = (lane & 1) * 16;     // A-tile load coords
    for (int k0 = 0; k0 < K; k0 += 32) {
        {   // A tile 16x32: each lane 16 contiguous floats = 4x b128
            const float4* src = (const float4*)(X + (size_t)(m0 + ar) * K + k0 + acb);
            float4* dst = (float4*)(sA + ar * 32 + acb);
#pragma unroll
            for (int j = 0; j < 4; ++j) dst[j] = src[j];
        }
        {   // B tile 32x16: each lane one K-row of 16 floats = 4x b128
            const float4* src = (const float4*)(W + (size_t)(k0 + lane) * Nn + n0);
            float4* dst = (float4*)(sB + lane * 16);
#pragma unroll
            for (int j = 0; j < 4; ++j) dst[j] = src[j];
        }
        if (k0 + 32 < K) {   // prefetch next tiles -> global_prefetch_b8
            __builtin_prefetch(X + (size_t)(m0 + ar) * K + k0 + 32 + acb, 0, 3);
            __builtin_prefetch(W + (size_t)(k0 + 32 + lane) * Nn + n0, 0, 3);
        }
        __syncthreads();
        v16h a, bf;
#pragma unroll
        for (int i = 0; i < 16; ++i) {
            int k = i + ((i >= 8) ? 8 : 0) + (hi ? 8 : 0);   // 16-bit A-frag map
            a[i] = (_Float16)sA[mr * 32 + k];
        }
#pragma unroll
        for (int i = 0; i < 16; ++i) bf[i] = (_Float16)sB[(i + ((i >= 8) ? 8 : 0) + (hi ? 8 : 0)) * 16 + mr]; // lane = N col, table-order K
        c = __builtin_amdgcn_wmma_f32_16x16x32_f16(false, a, false, bf, (short)0, c, false, false);
        __syncthreads();
    }
    const float bval = bias[n0 + mr];                     // hoisted: column-only
#pragma unroll
    for (int g = 0; g < 8; ++g) {
        int m = m0 + g + 8 * hi;
        float v = c[g] + bval;
        if (relu) v = fmaxf(v, 0.0f);
        Y[(size_t)m * Nn + n0 + mr] = v;
    }
}

// =====================================================================
// Fused flash-style attention with WMMA, one wave per (b, head, qtile, split).
// Q/Kb/Vb: [B, Nq|Nk, C] with head-packed channels (h*16+d).
// mode 0: self-attn, bias = 0 if qview==kview else -inf
// mode 1: cross-attn, bias = log-gaussian (view-gated, clamped at LOG_EPS)
// nsplit==1: write normalized Out directly. nsplit>1: write partial
// (unnormalized O, rowmax, rowsum) for attn_combine.
// =====================================================================
__global__ void attn_wmma(const float* __restrict__ Q, const float* __restrict__ Kb,
                          const float* __restrict__ Vb, float* __restrict__ Out,
                          const int* __restrict__ qview, const int* __restrict__ kview,
                          const float* __restrict__ centers, const float* __restrict__ inv2s2,
                          const float* __restrict__ dkpos,
                          float* __restrict__ pO, float* __restrict__ pM,
                          float* __restrict__ pL,
                          int Nq, int Nk, int mode, int nsplit) {
    const int lane = threadIdx.x;
    const int qt = blockIdx.x / nsplit, sp = blockIdx.x % nsplit;
    const int h = blockIdx.y, b = blockIdx.z;
    __shared__ float    sS[16][16];
    __shared__ _Float16 sP[16][32];   // K-padded probs (cols 16..31 stay 0)
    __shared__ float    sO[16][16];
    __shared__ float    rmax[16], rsum[16];
    __shared__ int      sKV[16];      // key views (self) 
    __shared__ float    sDK[16][2];   // key positions (cross)

    for (int i = lane; i < 16 * 32; i += 32) ((_Float16*)sP)[i] = (_Float16)0.0f;
    for (int i = lane; i < 256; i += 32) ((float*)sO)[i] = 0.0f;
    if (lane < 16) { rmax[lane] = NEG_; rsum[lane] = 0.0f; }

    const int mr = lane & 15, hi = lane >> 4;
    const int qn = qt * 16 + mr;

    // q fragment: head_dim=16 padded to K=32, pre-scaled by 1/sqrt(16)
    float qrow[16];
    {
        const float4* qsrc = (const float4*)(Q + ((size_t)(b * Nq + qn)) * C_ + h * 16);
#pragma unroll
        for (int j = 0; j < 4; ++j) ((float4*)qrow)[j] = qsrc[j];
    }
    v16h aq;
#pragma unroll
    for (int i = 0; i < 16; ++i) {
        int k = i + ((i >= 8) ? 8 : 0) + (hi ? 8 : 0);
        aq[i] = (_Float16)((k < 16) ? qrow[k] * 0.25f : 0.0f);
    }
    int   myview = 0; float cx = 0, cy = 0, is2 = 0;
    if (lane < 16) {
        int r = b * Nq + qt * 16 + lane;
        myview = qview[r];
        if (mode == 1) { cx = centers[r * 2]; cy = centers[r * 2 + 1]; is2 = inv2s2[r]; }
    }
    __syncthreads();

    const int ktiles = Nk / 16;
    const int ktn = ktiles / nsplit;
    const int ktbeg = sp * ktn, ktend = ktbeg + ktn;
    for (int kt = ktbeg; kt < ktend; ++kt) {
        const int kv0 = kt * 16;
        // KV_%16==0, so a tile never crosses a view boundary: tile-uniform view.
        const int vtile = kv0 / KV_;
        const int kvl0  = kv0 - vtile * KV_;
        // ---- S tile = q @ K^T ----
        v16h bk;
#pragma unroll
        for (int i = 0; i < 16; ++i) {
            int k = i + ((i >= 8) ? 8 : 0) + (hi ? 8 : 0);
            float kv = (k < 16) ? Kb[((size_t)(b * Nk + kv0 + mr)) * C_ + h * 16 + k] : 0.0f;
            bk[i] = (_Float16)kv;
        }
        if (kt + 1 < ktend) {
            __builtin_prefetch(Kb + ((size_t)(b * Nk + kv0 + 16 + mr)) * C_ + h * 16, 0, 3);
            __builtin_prefetch(Vb + ((size_t)(b * Nk + kv0 + 16 + mr)) * C_ + h * 16, 0, 3);
        }
        v8f cS = {};
        cS = __builtin_amdgcn_wmma_f32_16x16x32_f16(false, aq, false, bk, (short)0, cS, false, false);
#pragma unroll
        for (int g = 0; g < 8; ++g) sS[g + 8 * hi][mr] = cS[g];
        if (lane < 16) {
            if (mode == 0) sKV[lane] = kview[b * Nk + kv0 + lane];
            else { sDK[lane][0] = dkpos[(kvl0 + lane) * 2];
                   sDK[lane][1] = dkpos[(kvl0 + lane) * 2 + 1]; }
        }
        __syncthreads();

        // ---- online softmax (lanes 0..15, one row each) ----
        if (lane < 16) {
            const int r = lane;
            float sv[16];
            float tmax = NEG_;
#pragma unroll
            for (int j = 0; j < 16; ++j) {
                float bias;
                if (mode == 0) {
                    bias = (myview == sKV[j]) ? 0.0f : NEG_;
                } else if (myview != vtile) {
                    bias = NEG_;
                } else {
                    float dx = cx - sDK[j][0];
                    float dy = cy - sDK[j][1];
                    float lg = -(dx * dx + dy * dy) * is2;
                    bias = (lg < LOG_EPS_) ? NEG_ : lg;
                }
                float s = (bias <= -1.0e30f) ? NEG_ : (sS[r][j] + bias);
                sv[j] = s;
                tmax = fmaxf(tmax, s);
            }
            float nm = fmaxf(rmax[r], tmax);
            float f  = (rmax[r] <= -1.0e30f) ? ((nm <= -1.0e30f) ? 1.0f : 0.0f)
                                             : __expf(rmax[r] - nm);
            float add = 0.0f;
#pragma unroll
            for (int j = 0; j < 16; ++j) {
                float p = (sv[j] <= -1.0e30f) ? 0.0f : __expf(sv[j] - nm);
                sP[r][j] = (_Float16)p;
                add += p;
            }
            rsum[r] = rsum[r] * f + add;
            rmax[r] = nm;
#pragma unroll
            for (int j = 0; j < 16; ++j) sO[r][j] *= f;
        }
        __syncthreads();

        // ---- O += P @ V ----
        v16h ap;
#pragma unroll
        for (int i = 0; i < 16; ++i) {
            int k = i + ((i >= 8) ? 8 : 0) + (hi ? 8 : 0);
            ap[i] = sP[mr][k];                         // cols >=16 are zero
        }
        v16h bv;
        {
            float vrow[16];
            if (lane < 16) {
                const float4* vs = (const float4*)(Vb + ((size_t)(b * Nk + kv0 + lane)) * C_ + h * 16);
#pragma unroll
                for (int j = 0; j < 4; ++j) ((float4*)vrow)[j] = vs[j];
            } else {
#pragma unroll
                for (int j = 0; j < 16; ++j) vrow[j] = 0.0f;
            }
#pragma unroll
            for (int i = 0; i < 16; ++i) { int k = i + ((i >= 8) ? 8 : 0) + (hi ? 8 : 0); bv[i] = (_Float16)((k < 16) ? Vb[((size_t)(b * Nk + kv0 + k)) * C_ + h * 16 + mr] : 0.0f); }
        }
        v8f cO = {};
        cO = __builtin_amdgcn_wmma_f32_16x16x32_f16(false, ap, false, bv, (short)0, cO, false, false);
#pragma unroll
        for (int g = 0; g < 8; ++g) sO[g + 8 * hi][mr] += cO[g];
        __syncthreads();
    }

    if (nsplit == 1) {
#pragma unroll
        for (int g = 0; g < 8; ++g) {
            int r = g + 8 * hi;
            float den = rsum[r];
            Out[((size_t)(b * Nq + qt * 16 + r)) * C_ + h * 16 + mr] =
                sO[r][mr] / ((den > 0.0f) ? den : 1.0f);
        }
    } else {
        const int QT = Nq / 16;
        const size_t pb = (((size_t)(b * H_ + h) * QT + qt) * nsplit + sp);
#pragma unroll
        for (int g = 0; g < 8; ++g) {
            int r = g + 8 * hi;
            pO[pb * 256 + r * 16 + mr] = sO[r][mr];
        }
        if (lane < 16) { pM[pb * 16 + lane] = rmax[lane]; pL[pb * 16 + lane] = rsum[lane]; }
    }
}

// merge split-K flash partials: Out = sum_s e^(m_s-M) O_s / sum_s e^(m_s-M) l_s
__global__ void attn_combine(const float* __restrict__ pO, const float* __restrict__ pM,
                             const float* __restrict__ pL, float* __restrict__ Out,
                             int Nq, int nsplit) {
    const int qt = blockIdx.x, h = blockIdx.y, b = blockIdx.z;
    const int t = threadIdx.x, r = t >> 4, cc = t & 15;
    const int QT = Nq / 16;
    const size_t base = ((size_t)(b * H_ + h) * QT + qt) * nsplit;
    float M = NEG_;
    for (int s = 0; s < nsplit; ++s) M = fmaxf(M, pM[(base + s) * 16 + r]);
    float acc = 0.0f, L = 0.0f;
    for (int s = 0; s < nsplit; ++s) {
        float wgt = __expf(pM[(base + s) * 16 + r] - M);
        acc += wgt * pO[(base + s) * 256 + r * 16 + cc];
        L   += wgt * pL[(base + s) * 16 + r];
    }
    Out[((size_t)(b * Nq + qt * 16 + r)) * C_ + h * 16 + cc] = acc / ((L > 0.0f) ? L : 1.0f);
}

// ---------------- small scalar kernels ----------------

__global__ void mlp_first(const float* __restrict__ pos, const float* __restrict__ w1,
                          const float* __restrict__ b1, float* __restrict__ out, int rows) {
    int idx = blockIdx.x * blockDim.x + threadIdx.x;
    if (idx >= rows * C_) return;
    int r = idx / C_, c = idx - r * C_;
    float v = pos[r * 2] * w1[c] + pos[r * 2 + 1] * w1[C_ + c] + b1[c];
    out[idx] = fmaxf(v, 0.0f);
}

__global__ void denorm_kernel(const float* __restrict__ pos, float* __restrict__ dk) {
    int kv = blockIdx.x * blockDim.x + threadIdx.x;
    if (kv >= KV_) return;
    dk[kv * 2]     = sigm(pos[kv * 2])     * IMG_X_ - 0.5f;
    dk[kv * 2 + 1] = sigm(pos[kv * 2 + 1]) * IMG_Y_ - 0.5f;
}

__global__ void build_kk(const float* __restrict__ iff, const float* __restrict__ kpe,
                         float* __restrict__ kk) {
    size_t idx = (size_t)blockIdx.x * blockDim.x + threadIdx.x;
    if (idx >= (size_t)B_ * VK_ * C_) return;
    int c = (int)(idx % C_);
    int t = (int)((idx / C_) % VK_);
    int b = (int)(idx / ((size_t)C_ * VK_));
    int v = t / KV_, kv = t - v * KV_;
    kk[idx] = iff[(((size_t)(b * V_ + v)) * C_ + c) * KV_ + kv] + kpe[(size_t)kv * C_ + c];
}

__global__ void transpose_in(const float* __restrict__ iqf, float* __restrict__ feat) {
    int idx = blockIdx.x * blockDim.x + threadIdx.x;
    if (idx >= B_ * N_ * C_) return;
    int c = idx % C_, n = (idx / C_) % N_, b = idx / (C_ * N_);
    feat[idx] = iqf[(b * C_ + c) * N_ + n];
}

__global__ void transpose_out(const float* __restrict__ feat, float* __restrict__ out) {
    int idx = blockIdx.x * blockDim.x + threadIdx.x;
    if (idx >= B_ * C_ * N_) return;
    int n = idx % N_, c = (idx / N_) % C_, b = idx / (C_ * N_);
    out[idx] = feat[(b * N_ + n) * C_ + c];
}

__global__ void add_kernel(const float* __restrict__ x, const float* __restrict__ y,
                           float* __restrict__ z, int n) {
    int i = blockIdx.x * blockDim.x + threadIdx.x;
    if (i < n) z[i] = x[i] + y[i];
}

__global__ void copy_kernel(const float* __restrict__ x, float* __restrict__ y, int n) {
    int i = blockIdx.x * blockDim.x + threadIdx.x;
    if (i < n) y[i] = x[i];
}

__global__ void head2_kernel(const float* __restrict__ h1, const float* __restrict__ w2,
                             const float* __restrict__ b2, float* __restrict__ out) {
    int idx = blockIdx.x * blockDim.x + threadIdx.x;
    if (idx >= BN_ * 4) return;
    int bn = idx >> 2, j = idx & 3;
    float acc = b2[j];
    for (int k = 0; k < C_; ++k) acc += h1[bn * C_ + k] * w2[k * 4 + j];
    out[idx] = acc;
}

__global__ void bbox_init(const float* __restrict__ logits, const float* __restrict__ qpos0,
                          float* __restrict__ bstate) {
    int bn = blockIdx.x * blockDim.x + threadIdx.x;
    if (bn >= BN_) return;
    bstate[bn * 4]     = qpos0[bn * 2];
    bstate[bn * 4 + 1] = qpos0[bn * 2 + 1];
    bstate[bn * 4 + 2] = logits[bn * 4 + 2];
    bstate[bn * 4 + 3] = logits[bn * 4 + 3];
}

__global__ void geom_kernel(const float* __restrict__ qpos, const float* __restrict__ bstate,
                            float* __restrict__ centers, float* __restrict__ inv2s2) {
    int bn = blockIdx.x * blockDim.x + threadIdx.x;
    if (bn >= BN_) return;
    centers[bn * 2]     = sigm(qpos[bn * 2])     * IMG_X_;
    centers[bn * 2 + 1] = sigm(qpos[bn * 2 + 1]) * IMG_Y_;
    float dx = sigm(bstate[bn * 4 + 2]) * IMG_X_;
    float dy = sigm(bstate[bn * 4 + 3]) * IMG_Y_;
    float radius = ceilf(sqrtf(dx * dx + dy * dy) * 0.5f);
    float sg = (radius * 2.0f + 1.0f) / 6.0f;
    inv2s2[bn] = 1.0f / (2.0f * sg * sg);
}

__global__ void head_finalize(const float* __restrict__ logits, const float* __restrict__ qcur,
                              float* __restrict__ qnext, float* __restrict__ bstate,
                              float* __restrict__ ret) {
    int bn = blockIdx.x * blockDim.x + threadIdx.x;
    if (bn >= BN_) return;
    int b = bn / N_, n = bn - b * N_;
#pragma unroll
    for (int j = 0; j < 4; ++j) {
        float v = logits[bn * 4 + j] + ((j < 2) ? qcur[bn * 2 + j] : 0.0f);
        if (j < 2) qnext[bn * 2 + j] = v;
        float s = sigm(v);
        bstate[bn * 4 + j] = s;
        ret[(b * 4 + j) * N_ + n] = s;
    }
}

__global__ void ln_residual(const float* __restrict__ x, const float* __restrict__ dx,
                            const float* __restrict__ g, const float* __restrict__ bb,
                            float* __restrict__ y) {
    int r = blockIdx.x, t = threadIdx.x;
    __shared__ float red[C_];
    float v = x[(size_t)r * C_ + t] + dx[(size_t)r * C_ + t];
    red[t] = v; __syncthreads();
    for (int s = C_ / 2; s > 0; s >>= 1) { if (t < s) red[t] += red[t + s]; __syncthreads(); }
    float mean = red[0] / C_;
    __syncthreads();
    float d = v - mean;
    red[t] = d * d; __syncthreads();
    for (int s = C_ / 2; s > 0; s >>= 1) { if (t < s) red[t] += red[t + s]; __syncthreads(); }
    float var = red[0] / C_;
    y[(size_t)r * C_ + t] = d * rsqrtf(var + 1e-5f) * g[t] + bb[t];
}

// =====================================================================
// Host orchestration
// =====================================================================
extern "C" void kernel_launch(void* const* d_in, const int* in_sizes, int n_in,
                              void* d_out, int out_size, void* d_ws, size_t ws_size,
                              hipStream_t stream) {
    (void)in_sizes; (void)out_size; (void)ws_size;

    const float* iqf  = (const float*)d_in[0];   // [B,C,N]
    const float* qpos0 = (const float*)d_in[1];  // [B,N,2]
    const float* iff  = (const float*)d_in[2];   // [B,V,C,KV]
    const float* kpos_in = (const float*)d_in[3];// [1,KV,2]

    // ---- parameter leaves (dict insertion order) ----
    const float* leaf[64];
    const int* qview;
    if (n_in >= 69) {
        for (int i = 0; i < 64; ++i) leaf[i] = (const float*)d_in[4 + i];
        qview = (const int*)d_in[68];
    } else {
        static const int ls[64] = {
            256,128,16384,128, 256,128,16384,128, 16384,128,512,4,
            16384,128,16384,128,16384,128,16384,128, 16384,128,16384,128,16384,128,16384,128,
            32768,256,32768,128, 128,128,128,128,128,128,
            16384,128,16384,128,16384,128,16384,128, 16384,128,16384,128,16384,128,16384,128,
            32768,256,32768,128, 128,128,128,128,128,128 };
        const float* base = (const float*)d_in[4];
        size_t o = 0;
        for (int i = 0; i < 64; ++i) { leaf[i] = base + o; o += (size_t)ls[i]; }
        qview = (const int*)d_in[5];
    }
    const float *qpos_w1 = leaf[0], *qpos_b1 = leaf[1], *qpos_w2 = leaf[2], *qpos_b2 = leaf[3];
    const float *kpos_w1 = leaf[4], *kpos_b1 = leaf[5], *kpos_w2 = leaf[6], *kpos_b2 = leaf[7];
    const float *head_w1 = leaf[8], *head_b1 = leaf[9], *head_w2 = leaf[10], *head_b2 = leaf[11];

    // ---- workspace carve-up (floats) ----
    float* w = (float*)d_ws;
    size_t off = 0;
    auto alloc = [&](size_t n) { float* p = w + off; off += n; return p; };
    float* kpe    = alloc((size_t)KV_ * C_);
    float* dkpos  = alloc((size_t)KV_ * 2);
    float* kk     = alloc((size_t)B_ * VK_ * C_);
    float* kh     = alloc((size_t)B_ * VK_ * C_);
    float* vvh    = alloc((size_t)B_ * VK_ * C_);
    float* feat   = alloc((size_t)BN_ * C_);
    float* qpe    = alloc((size_t)BN_ * C_);
    float* xq     = alloc((size_t)BN_ * C_);
    float* sq     = alloc((size_t)BN_ * C_);
    float* skb    = alloc((size_t)BN_ * C_);
    float* svb    = alloc((size_t)BN_ * C_);
    float* t0     = alloc((size_t)BN_ * C_);
    float* t1     = alloc((size_t)BN_ * C_);
    float* hid256 = alloc((size_t)BN_ * FFN_);
    float* hidK   = alloc((size_t)KV_ * C_);
    float* hidQ   = alloc((size_t)BN_ * C_);
    float* h1     = alloc((size_t)BN_ * C_);
    float* logits4= alloc((size_t)BN_ * 4);
    float* bstate = alloc((size_t)BN_ * 4);
    float* qnA    = alloc((size_t)BN_ * 2);
    float* qnB    = alloc((size_t)BN_ * 2);
    float* centers= alloc((size_t)BN_ * 2);
    float* i2s2   = alloc((size_t)BN_);
    float* pO     = alloc((size_t)B_ * H_ * QT_ * SPLIT_ * 256);
    float* pM     = alloc((size_t)B_ * H_ * QT_ * SPLIT_ * 16);
    float* pL     = alloc((size_t)B_ * H_ * QT_ * SPLIT_ * 16);

    float* out = (float*)d_out;
    const size_t OUT_QPOS = (size_t)B_ * C_ * N_;          // 102400
    const size_t OUT_RET0 = OUT_QPOS + (size_t)BN_ * 2;    // 104000
    const size_t OUT_RET1 = OUT_RET0 + (size_t)B_ * 4 * N_;// 107200

    auto cdiv = [](int a, int b) { return (a + b - 1) / b; };

    // ---- stage 0: key positional embedding + key tensor ----
    mlp_first<<<cdiv(KV_ * C_, 256), 256, 0, stream>>>(kpos_in, kpos_w1, kpos_b1, hidK, KV_);
    wmma_gemm<<<dim3(KV_ / 16, C_ / 16), 32, 0, stream>>>(hidK, kpos_w2, kpos_b2, kpe, KV_, C_, C_, 0);
    denorm_kernel<<<cdiv(KV_, 256), 256, 0, stream>>>(kpos_in, dkpos);
    build_kk<<<cdiv(B_ * VK_ * C_, 256), 256, 0, stream>>>(iff, kpe, kk);
    transpose_in<<<cdiv(B_ * N_ * C_, 256), 256, 0, stream>>>(iqf, feat);

    // ---- initial bbox head ----
    wmma_gemm<<<dim3(BN_ / 16, C_ / 16), 32, 0, stream>>>(feat, head_w1, head_b1, h1, BN_, C_, C_, 1);
    head2_kernel<<<cdiv(BN_ * 4, 256), 256, 0, stream>>>(h1, head_w2, head_b2, logits4);
    bbox_init<<<cdiv(BN_, 256), 256, 0, stream>>>(logits4, qpos0, bstate);

    const float* qcur = qpos0;
    for (int i = 0; i < 2; ++i) {
        const float* const* L = leaf + 12 + 26 * i;
        const float *sa_wq = L[0],  *sa_bq = L[1],  *sa_wk = L[2],  *sa_bk = L[3];
        const float *sa_wv = L[4],  *sa_bv = L[5],  *sa_wo = L[6],  *sa_bo = L[7];
        const float *ca_wq = L[8],  *ca_bq = L[9],  *ca_wk = L[10], *ca_bk = L[11];
        const float *ca_wv = L[12], *ca_bv = L[13], *ca_wo = L[14], *ca_bo = L[15];
        const float *f1w = L[16], *f1b = L[17], *f2w = L[18], *f2b = L[19];
        const float *n1g = L[20], *n1b = L[21], *n2g = L[22], *n2b = L[23], *n3g = L[24], *n3b = L[25];

        // geometry + query pos embedding
        geom_kernel<<<cdiv(BN_, 256), 256, 0, stream>>>(qcur, bstate, centers, i2s2);
        mlp_first<<<cdiv(BN_ * C_, 256), 256, 0, stream>>>(qcur, qpos_w1, qpos_b1, hidQ, BN_);
        wmma_gemm<<<dim3(BN_ / 16, C_ / 16), 32, 0, stream>>>(hidQ, qpos_w2, qpos_b2, qpe, BN_, C_, C_, 0);

        // cross K/V projections over all 33792 keys (largest GEMMs)
        wmma_gemm<<<dim3(B_ * VK_ / 16, C_ / 16), 32, 0, stream>>>(kk, ca_wk, ca_bk, kh,  B_ * VK_, C_, C_, 0);
        wmma_gemm<<<dim3(B_ * VK_ / 16, C_ / 16), 32, 0, stream>>>(kk, ca_wv, ca_bv, vvh, B_ * VK_, C_, C_, 0);

        // ---- self attention ----
        add_kernel<<<cdiv(BN_ * C_, 256), 256, 0, stream>>>(feat, qpe, xq, BN_ * C_);
        wmma_gemm<<<dim3(BN_ / 16, C_ / 16), 32, 0, stream>>>(xq, sa_wq, sa_bq, sq,  BN_, C_, C_, 0);
        wmma_gemm<<<dim3(BN_ / 16, C_ / 16), 32, 0, stream>>>(xq, sa_wk, sa_bk, skb, BN_, C_, C_, 0);
        wmma_gemm<<<dim3(BN_ / 16, C_ / 16), 32, 0, stream>>>(xq, sa_wv, sa_bv, svb, BN_, C_, C_, 0);
        attn_wmma<<<dim3(QT_, H_, B_), 32, 0, stream>>>(sq, skb, svb, t0, qview, qview,
                                                        nullptr, nullptr, nullptr,
                                                        nullptr, nullptr, nullptr, N_, N_, 0, 1);
        wmma_gemm<<<dim3(BN_ / 16, C_ / 16), 32, 0, stream>>>(t0, sa_wo, sa_bo, t1, BN_, C_, C_, 0);
        ln_residual<<<BN_, C_, 0, stream>>>(feat, t1, n1g, n1b, feat);

        // ---- cross attention (split-K flash + combine) ----
        add_kernel<<<cdiv(BN_ * C_, 256), 256, 0, stream>>>(feat, qpe, xq, BN_ * C_);
        wmma_gemm<<<dim3(BN_ / 16, C_ / 16), 32, 0, stream>>>(xq, ca_wq, ca_bq, sq, BN_, C_, C_, 0);
        attn_wmma<<<dim3(QT_ * SPLIT_, H_, B_), 32, 0, stream>>>(sq, kh, vvh, nullptr, qview, nullptr,
                                                                 centers, i2s2, dkpos,
                                                                 pO, pM, pL, N_, VK_, 1, SPLIT_);
        attn_combine<<<dim3(QT_, H_, B_), 256, 0, stream>>>(pO, pM, pL, t0, N_, SPLIT_);
        wmma_gemm<<<dim3(BN_ / 16, C_ / 16), 32, 0, stream>>>(t0, ca_wo, ca_bo, t1, BN_, C_, C_, 0);
        ln_residual<<<BN_, C_, 0, stream>>>(feat, t1, n2g, n2b, feat);

        // ---- FFN ----
        wmma_gemm<<<dim3(BN_ / 16, FFN_ / 16), 32, 0, stream>>>(feat, f1w, f1b, hid256, BN_, C_, FFN_, 1);
        wmma_gemm<<<dim3(BN_ / 16, C_ / 16), 32, 0, stream>>>(hid256, f2w, f2b, t1, BN_, FFN_, C_, 0);
        ln_residual<<<BN_, C_, 0, stream>>>(feat, t1, n3g, n3b, feat);

        // ---- bbox head + state update ----
        wmma_gemm<<<dim3(BN_ / 16, C_ / 16), 32, 0, stream>>>(feat, head_w1, head_b1, h1, BN_, C_, C_, 1);
        head2_kernel<<<cdiv(BN_ * 4, 256), 256, 0, stream>>>(h1, head_w2, head_b2, logits4);
        float* qnext = (i == 0) ? qnA : qnB;
        head_finalize<<<cdiv(BN_, 256), 256, 0, stream>>>(logits4, qcur, qnext, bstate,
                                                          out + ((i == 0) ? OUT_RET0 : OUT_RET1));
        qcur = qnext;
    }

    // ---- final outputs: feat^T [B,C,N] and final qpos [B,N,2] ----
    transpose_out<<<cdiv(B_ * C_ * N_, 256), 256, 0, stream>>>(feat, out);
    copy_kernel<<<cdiv(BN_ * 2, 256), 256, 0, stream>>>(qcur, out + OUT_QPOS, BN_ * 2);
}
